// CannyFilter_77129022701842
// MI455X (gfx1250) — matrix operand order repaired and checked by reading, not compiled, not run on hardware
//
#include <hip/hip_runtime.h>
#include <math.h>

// ---------------------------------------------------------------------------
// Fully fused Canny pipeline for MI455X (gfx1250, wave32).
//   m  = mean_c(img)                      (40x40 tile+halo4, masked to image)
//   b  = gauss (3x3) * m                  (38x38)   WMMA stage, 1 filter
//   gm = sqrt((sx*b)^2 + (sy*b)^2)        (36x36)   WMMA stage, 2 filters
//   S  = (sum_d dir_w[d]) (3x3) * gm      (34x34)   WMMA stage, 1 filter
//   out= nms_row (3x3) * S                (32x32)   WMMA stage, 1 filter
// Every intermediate is zero-masked outside the image to reproduce the
// crop-then-zero-pad semantics of each conv in the reference.
// Image tile -> LDS uses GLOBAL_LOAD_ASYNC_TO_LDS_B32 when the toolchain
// exposes it (ASYNCcnt-tracked DMA, no VGPR staging), else plain loads.
// ---------------------------------------------------------------------------

typedef __attribute__((ext_vector_type(16))) _Float16 v16h;
typedef __attribute__((ext_vector_type(8)))  float    v8f;

typedef __attribute__((address_space(1))) int g_int;
typedef __attribute__((address_space(3))) int l_int;

#define IMG_H 512
#define IMG_W 512
#define NCH   3
#define TILE  32
#define MD    40   // mean buffer (halo 4)
#define BD    38   // blurred     (halo 3)
#define GD    36   // grad mag    (halo 2)
#define SD    34   // dir-sum S   (halo 1)
#define NWAVES 8

#if defined(__has_builtin)
#if __has_builtin(__builtin_amdgcn_global_load_async_to_lds_b32) && \
    __has_builtin(__builtin_amdgcn_s_wait_asynccnt)
#define USE_ASYNC_LDS 1
#endif
#endif
#ifndef USE_ASYNC_LDS
#define USE_ASYNC_LDS 0
#endif

// One 3x3 conv stage over an OD x OD region, input IP x IP (IP = OD+2),
// computed as im2col GEMM on v_wmma_f32_16x16x32_f16:
//   A (16x32 f16): M = filter index (NF used), K = 9 taps (rest zero)
//   B (32x16 f16): N = pixel within 16-pixel group, K = 9 patch taps
//   D (16x16 f32): lane n (n<16), VGPR f  ->  filter f at pixel n  (in-lane!)
template<int IP, int OD, int NF, int POST, bool TOGLOBAL>
__device__ __forceinline__ void conv3x3_stage(const float* __restrict__ inb,
                                              float* __restrict__ outb,
                                              const float* __restrict__ flds,
                                              int gy0, int gx0,
                                              float* __restrict__ gout, int bz,
                                              int lane, int wave)
{
  // Build A once per stage (constant across pixel groups).
  // 16-bit A layout (contiguous K): lanes 0-15 hold K=0..15 in elements
  // 0..15; lanes 16-31 hold K=16..31. Only K<9 is nonzero.
  v16h a = {};
  const int f  = lane & 15;
  const int kb = (lane >> 4) * 16;
#pragma unroll
  for (int e = 0; e < 9; ++e) {
    const int K = kb + e;
    if (f < NF && K < 9) a[e] = (_Float16)flds[f * 9 + K];
  }

  const int n  = OD * OD;
  const int ng = (n + 15) >> 4;          // 16-pixel groups
  for (int g = wave; g < ng; g += NWAVES) {   // uniform per wave
    const int  p     = g * 16 + f;
    const bool valid = (lane < 16) && (p < n);
    const int  pp    = (p < n) ? p : 0;
    const int  by    = pp / OD;
    const int  bx    = pp - by * OD;

    // B: lane n<16 supplies column n = 9-tap patch of pixel p (K rows 0..8);
    // lanes 16-31 cover K=16..31 which are all zero.
    v16h b = {};
    if (valid) {
#pragma unroll
      for (int t = 0; t < 9; ++t)
        b[t] = (_Float16)inb[(by + t / 3) * IP + (bx + t % 3)];
    }

    v8f c = {};
    c = __builtin_amdgcn_wmma_f32_16x16x32_f16(false, a, false, b, (short)0,
                                               c, false, false);

    if (valid) {
      float r = c[0];                       // filter 0 for this pixel
      if (POST) r = sqrtf(c[0] * c[0] + c[1] * c[1]);  // sobel magnitude
      const int  gy    = gy0 + by, gx = gx0 + bx;
      const bool inimg = ((unsigned)gy < IMG_H) && ((unsigned)gx < IMG_W);
      if (TOGLOBAL) {
        if (inimg) gout[((size_t)bz * IMG_H + gy) * IMG_W + gx] = r;
      } else {
        outb[p] = inimg ? r : 0.0f;         // reproduce crop/zero-pad
      }
    }
  }
}

__global__ __launch_bounds__(256) void canny_fused_kernel(
    const float* __restrict__ img, const float* __restrict__ gw,
    const float* __restrict__ sxw, const float* __restrict__ syw,
    const float* __restrict__ dirw, const float* __restrict__ nmsw,
    float* __restrict__ out)
{
  __shared__ float s_m[MD * MD];
  __shared__ float s_b[BD * BD];
  __shared__ float s_g[GD * GD];
  __shared__ float s_s[SD * SD];
  __shared__ float s_f[4 * 18];   // stage filter banks: [stage][filter][9]
#if USE_ASYNC_LDS
  __shared__ float s_c[NCH][MD * MD];   // raw channel staging (async DMA dst)
#endif

  const int tid  = threadIdx.x;
  const int lane = tid & 31;
  const int wave = tid >> 5;
  const int bz   = blockIdx.z;
  const int ty0  = blockIdx.y * TILE;
  const int tx0  = blockIdx.x * TILE;
  const float* imgb = img + (size_t)bz * NCH * IMG_H * IMG_W;

#if USE_ASYNC_LDS
  // Kick off async global->LDS copies of the 3 channel tiles first so the
  // DMA overlaps with the filter-bank setup below. Out-of-image halo lanes
  // clamp their *addresses* into the buffer (their values are masked to zero
  // in the reduction pass, but the copy itself must not touch OOB memory).
  for (int i = tid; i < MD * MD; i += 256) {
    const int y  = i / MD, x = i - y * MD;
    int gy = ty0 - 4 + y, gx = tx0 - 4 + x;
    gy = gy < 0 ? 0 : (gy > IMG_H - 1 ? IMG_H - 1 : gy);
    gx = gx < 0 ? 0 : (gx > IMG_W - 1 ? IMG_W - 1 : gx);
    const size_t o = (size_t)gy * IMG_W + gx;
#pragma unroll
    for (int c = 0; c < NCH; ++c) {
      __builtin_amdgcn_global_load_async_to_lds_b32(
          (g_int*)(imgb + o + (size_t)c * IMG_H * IMG_W),
          (l_int*)&s_c[c][i],
          0, 0);
    }
  }
#endif

  // Effective filters, computed from the actual input weight tensors.
  if (tid < 9) {
    s_f[0 * 18 + tid]     = gw[tid];     // gaussian (channel 0 of (C,1,3,3))
    s_f[1 * 18 + tid]     = sxw[tid];    // sobel x
    s_f[1 * 18 + 9 + tid] = syw[tid];    // sobel y
    float ws = 0.0f;                     // sum of 8 direction kernels
    for (int d = 0; d < 8; ++d) ws += dirw[d * 9 + tid];
    s_f[2 * 18 + tid] = ws;
    s_f[3 * 18 + tid] = nmsw[tid];       // nms taps (identical per channel)
  }

#if USE_ASYNC_LDS
  __builtin_amdgcn_s_wait_asynccnt(0);   // this wave's DMAs landed in LDS
  __syncthreads();                       // all waves' DMAs visible
  // Reduce staged channels to the masked mean buffer.
  for (int i = tid; i < MD * MD; i += 256) {
    const int y  = i / MD, x = i - y * MD;
    const int gy = ty0 - 4 + y, gx = tx0 - 4 + x;
    float v = 0.0f;
    if ((unsigned)gy < IMG_H && (unsigned)gx < IMG_W)
      v = (s_c[0][i] + s_c[1][i] + s_c[2][i]) * (1.0f / 3.0f);
    s_m[i] = v;
  }
#else
  // Fallback: channel mean with halo 4, zero outside the image.
  for (int i = tid; i < MD * MD; i += 256) {
    const int y  = i / MD, x = i - y * MD;
    const int gy = ty0 - 4 + y, gx = tx0 - 4 + x;
    float v = 0.0f;
    if ((unsigned)gy < IMG_H && (unsigned)gx < IMG_W) {
      const size_t o = (size_t)gy * IMG_W + gx;
      v = (imgb[o] + imgb[o + IMG_H * IMG_W] + imgb[o + 2 * IMG_H * IMG_W]) *
          (1.0f / 3.0f);
    }
    s_m[i] = v;
  }
#endif
  __syncthreads();

  conv3x3_stage<MD, BD, 1, 0, false>(s_m, s_b, &s_f[0 * 18], ty0 - 3, tx0 - 3,
                                     nullptr, bz, lane, wave);
  __syncthreads();
  conv3x3_stage<BD, GD, 2, 1, false>(s_b, s_g, &s_f[1 * 18], ty0 - 2, tx0 - 2,
                                     nullptr, bz, lane, wave);
  __syncthreads();
  conv3x3_stage<GD, SD, 1, 0, false>(s_g, s_s, &s_f[2 * 18], ty0 - 1, tx0 - 1,
                                     nullptr, bz, lane, wave);
  __syncthreads();
  conv3x3_stage<SD, TILE, 1, 0, true>(s_s, nullptr, &s_f[3 * 18], ty0, tx0,
                                      out, bz, lane, wave);
}

extern "C" void kernel_launch(void* const* d_in, const int* in_sizes, int n_in,
                              void* d_out, int out_size, void* d_ws,
                              size_t ws_size, hipStream_t stream) {
  const float* img  = (const float*)d_in[0];
  const float* gw   = (const float*)d_in[1];
  const float* sxw  = (const float*)d_in[2];
  const float* syw  = (const float*)d_in[3];
  const float* dirw = (const float*)d_in[4];
  const float* nmsw = (const float*)d_in[5];
  float* out = (float*)d_out;

  dim3 grid(IMG_W / TILE, IMG_H / TILE, 32);
  canny_fused_kernel<<<grid, dim3(256), 0, stream>>>(img, gw, sxw, syw, dirw,
                                                     nmsw, out);
}
